// AttentionBlock_21431886807687
// MI455X (gfx1250) — matrix operand order repaired and checked by reading, not compiled, not run on hardware
//
#include <hip/hip_runtime.h>
#include <hip/hip_bf16.h>

typedef __attribute__((ext_vector_type(16))) _Float16 v16h;
typedef __attribute__((ext_vector_type(8)))  _Float16 v8h;
typedef __attribute__((ext_vector_type(8)))  float    v8f;

#define B_      8
#define C_      512
#define T_      1024
#define HEADS_  8
#define CH_     64          // C/HEADS
#define GROUPS_ 32
#define CPG_    16          // C/GROUPS
#define O3C_    1536        // 3*C

// ---------------- workspace layout (bytes) ----------------
#define XN_OFF  ((size_t)0)                       // f16 [B][T][C]     8 MiB  (t-major, c contiguous)
#define QT_OFF  ((size_t)8388608)                 // f16 [BH][T][64]   8 MiB  (scaled)
#define KT_OFF  ((size_t)16777216)                // f16 [BH][T][64]   8 MiB  (scaled)
#define VT_OFF  ((size_t)25165824)                // f16 [BH][64][T]   8 MiB  (c-major, s contiguous)
#define AH_OFF  ((size_t)33554432)                // f16 [B][T][C]     8 MiB  (t-major, c contiguous)
#define WQH_OFF ((size_t)41943040)                // f16 [1536][512]
#define WPH_OFF ((size_t)43515904)                // f16 [512][512]

// ------------- WMMA fragment loaders (ISA 7.12.2, wave32), 2x b128 each ----
// A 16x32 f16 from row-major [M][K] (rs = row stride in elems).
// lane l: row = l&15, K runs [kb, kb+8) and [kb+16, kb+24), kb = 8*(l>>4).
__device__ inline v16h load_fA_k(const _Float16* p0, int rs, int lane) {
    const int r  = lane & 15;
    const int kb = (lane >> 4) * 8;
    const _Float16* p = p0 + (size_t)r * rs + kb;
    union { v16h v; v8h h[2]; } u;
    u.h[0] = *(const v8h*)(p);        // K = kb .. kb+7   -> vgprs 0..3
    u.h[1] = *(const v8h*)(p + 16);   // K = kb+16..kb+23 -> vgprs 4..7
    return u.v;
}

// B 32x16 f16 from transposed [N][K] (ns = column stride in elems).
// lane l: col = l&15, K runs [kb, kb+8) and [kb+16, kb+24), kb = 8*(l>>4).
__device__ inline v16h load_fB_k(const _Float16* p0, int ns, int lane) {
    const int n  = lane & 15;
    const int kb = (lane >> 4) * 8;
    const _Float16* p = p0 + (size_t)n * ns + kb;
    union { v16h v; v8h h[2]; } u;
    u.h[0] = *(const v8h*)(p);
    u.h[1] = *(const v8h*)(p + 16);
    return u.v;
}

__device__ inline v8f wmma16x16x32(v16h a, v16h b, v8f c) {
    return __builtin_amdgcn_wmma_f32_16x16x32_f16(
        false, a, false, b, (short)0, c, false, false);
}

// ---- CDNA5 async global->LDS copy (ASYNCcnt path, ISA 10.4 / 15.18.3) ----
// Per-lane: LDS[lds_addr] = MEM[gptr], 16 bytes, no VGPR bounce.
__device__ inline void async_copy_b128(unsigned lds_addr, const void* gptr) {
    asm volatile("global_load_async_to_lds_b128 %0, %1, off"
                 :: "v"(lds_addr), "v"(gptr) : "memory");
}
__device__ inline void wait_asynccnt0() {
    asm volatile("s_wait_asynccnt 0x0" ::: "memory");
}

// ---------------- kernel 0: weight f32 -> f16 ----------------
__global__ __launch_bounds__(256) void cvt_weights(const float* __restrict__ wq,
                                                   const float* __restrict__ wp,
                                                   _Float16* __restrict__ wqh,
                                                   _Float16* __restrict__ wph) {
    const int i = blockIdx.x * 256 + threadIdx.x;
    if (i < O3C_ * C_) wqh[i] = (_Float16)wq[i];
    if (i < C_ * C_)   wph[i] = (_Float16)wp[i];
}

// ------------- kernel 1: GroupNorm -> f16 xn[b][t][c] (LDS transpose) -------
__global__ __launch_bounds__(256) void groupnorm(const float* __restrict__ x,
                                                 const float* __restrict__ gw,
                                                 const float* __restrict__ gb,
                                                 _Float16* __restrict__ xn) {
    __shared__ float s1[256], s2[256];
    __shared__ __align__(16) _Float16 tile[T_][CPG_];     // 32 KiB, [t][c_local]
    const int tid = threadIdx.x;
    const int bg  = blockIdx.x;                 // b*32 + g
    const int g   = bg & (GROUPS_ - 1);
    const int b   = bg >> 5;
    const size_t base = (size_t)bg * (CPG_ * T_);         // contiguous 16ch x 1024t
    float sum = 0.f, sq = 0.f;
    for (int i = tid; i < CPG_ * T_; i += 256) {
        const float v = x[base + i];
        sum += v; sq += v * v;
    }
    s1[tid] = sum; s2[tid] = sq;
    __syncthreads();
    for (int st = 128; st > 0; st >>= 1) {
        if (tid < st) { s1[tid] += s1[tid + st]; s2[tid] += s2[tid + st]; }
        __syncthreads();
    }
    const float inv = 1.0f / (float)(CPG_ * T_);
    const float mu  = s1[0] * inv;
    const float var = s2[0] * inv - mu * mu;
    const float rs  = rsqrtf(var + 1e-5f);
    for (int i = tid; i < CPG_ * T_; i += 256) {
        const int cl = i >> 10;                  // local channel
        const int t  = i & (T_ - 1);
        const int c  = g * CPG_ + cl;
        tile[t][cl] = (_Float16)((x[base + i] - mu) * rs * gw[c] + gb[c]);
    }
    __syncthreads();
    // write out transposed, dword-coalesced: 16 f16 per t = 8 dwords
    for (int i = tid; i < T_ * (CPG_ / 2); i += 256) {
        const int t = i >> 3, d = i & 7;
        unsigned int* dst = (unsigned int*)(xn + ((size_t)(b * T_ + t)) * C_ + g * CPG_);
        dst[d] = ((const unsigned int*)tile)[t * 8 + d];
    }
}

// ------------- kernel 2: QKV GEMM (WMMA) + head-split scatter --------------
// qkv[o,t] = wq[o,:].xn[t,:] + bias[o];  o = h*192 + r; r<64 q, <128 k, else v
__global__ __launch_bounds__(128) void qkv_gemm(const _Float16* __restrict__ xn,
                                                const _Float16* __restrict__ wq,
                                                const float* __restrict__ bias,
                                                _Float16* __restrict__ qt,
                                                _Float16* __restrict__ kt,
                                                _Float16* __restrict__ vt) {
    const int lane = threadIdx.x & 31;
    const int wv   = threadIdx.x >> 5;
    const int b    = blockIdx.z;
    const int m0   = blockIdx.y * 64 + wv * 16;     // o base (0..1535)
    const int n0   = blockIdx.x * 64;               // t base
    const _Float16* A  = wq + (size_t)m0 * C_;                  // [o][c]
    const _Float16* Bp = xn + ((size_t)b * T_ + n0) * C_;       // [t][c]
    v8f acc[4] = {};
#pragma unroll 4
    for (int k = 0; k < C_; k += 32) {
        const v16h a = load_fA_k(A + k, C_, lane);
#pragma unroll
        for (int nt = 0; nt < 4; ++nt) {
            const v16h bb = load_fB_k(Bp + (size_t)(nt * 16) * C_ + k, C_, lane);
            acc[nt] = wmma16x16x32(a, bb, acc[nt]);
        }
    }
    const float scale = 0.3535533905932738f;   // 1/sqrt(sqrt(64))
    const int hl = lane >> 4, ln = lane & 15;
#pragma unroll
    for (int nt = 0; nt < 4; ++nt)
#pragma unroll
        for (int j = 0; j < 8; ++j) {
            const int o = m0 + j + 8 * hl;
            const int t = n0 + nt * 16 + ln;
            const float val = acc[nt][j] + bias[o];
            const int h = o / 192, r = o % 192;
            const int bh = b * HEADS_ + h;
            if (r < 64)
                qt[((size_t)bh * T_ + t) * CH_ + r] = (_Float16)(val * scale);
            else if (r < 128)
                kt[((size_t)bh * T_ + t) * CH_ + (r - 64)] = (_Float16)(val * scale);
            else
                vt[((size_t)bh * CH_ + (r - 128)) * T_ + t] = (_Float16)val;
        }
}

// ------------- kernel 3: fused flash attention (async-LDS-staged K/V) ------
__global__ __launch_bounds__(128) void flash_attn(const _Float16* __restrict__ qt,
                                                  const _Float16* __restrict__ kt,
                                                  const _Float16* __restrict__ vt,
                                                  _Float16* __restrict__ ah) {
    __shared__ __align__(16) _Float16 Kt[128][CH_];     // 16 KiB [s_local][c]
    __shared__ __align__(16) _Float16 Vt[CH_][128];     // 16 KiB [c][s_local]
    __shared__ __align__(16) _Float16 Pb[4][16][128];   // 16 KiB per-wave P
    const int tid  = threadIdx.x;
    const int lane = tid & 31;
    const int wv   = tid >> 5;
    const int bh   = blockIdx.y;
    const int m0   = blockIdx.x * 64 + wv * 16;         // query-row base
    const int hl   = lane >> 4, ln = lane & 15;

    // LDS byte addresses (flat->LDS truncation, ISA 10.2)
    const unsigned kb_lds = (unsigned)(uintptr_t)&Kt[0][0];
    const unsigned vb_lds = (unsigned)(uintptr_t)&Vt[0][0];

    // A = q^T tile: qt[bh][t][c], K=c contiguous
    const _Float16* qp = qt + ((size_t)bh * T_ + m0) * CH_;
    const v16h aq0 = load_fA_k(qp,      CH_, lane);
    const v16h aq1 = load_fA_k(qp + 32, CH_, lane);

    v8f accO[4] = {};
    float mi[8], li[8];
#pragma unroll
    for (int j = 0; j < 8; ++j) { mi[j] = -1e30f; li[j] = 0.f; }

    for (int s0 = 0; s0 < T_; s0 += 128) {
        // ---- async b128 stage of K (128x64, contiguous) and V (64x128) ----
        const _Float16* gk = kt + ((size_t)bh * T_ + s0) * CH_;   // 16 KiB
        const _Float16* gv = vt + (size_t)bh * CH_ * T_ + s0;     // 64 rows x 256B
#pragma unroll
        for (int r = 0; r < 8; ++r) {
            const int i = tid + r * 128;                          // 0..1023
            async_copy_b128(kb_lds + i * 16, gk + i * 8);
            const int c = i >> 4, d = i & 15;
            async_copy_b128(vb_lds + i * 16, gv + (size_t)c * T_ + d * 8);
        }
        wait_asynccnt0();
        __syncthreads();

        // ---- S = q^T k : 16 x 128 scores, f32 ----
        v8f S[8];
#pragma unroll
        for (int st = 0; st < 8; ++st) {
            const v16h b0 = load_fB_k(&Kt[st * 16][0],  CH_, lane);
            const v16h b1 = load_fB_k(&Kt[st * 16][32], CH_, lane);
            v8f s = {};
            s = wmma16x16x32(aq0, b0, s);
            s = wmma16x16x32(aq1, b1, s);
            S[st] = s;
        }

        // ---- online softmax (lane = col, vgpr j = row) ----
#pragma unroll
        for (int j = 0; j < 8; ++j) {
            float mx = mi[j];
#pragma unroll
            for (int st = 0; st < 8; ++st) mx = fmaxf(mx, S[st][j]);
#pragma unroll
            for (int d = 1; d < 16; d <<= 1) mx = fmaxf(mx, __shfl_xor(mx, d, 32));
            const float alpha = __expf(mi[j] - mx);
            mi[j] = mx;
            float rsum = 0.f;
#pragma unroll
            for (int st = 0; st < 8; ++st) {
                const float e = __expf(S[st][j] - mx);
                S[st][j] = e;
                rsum += e;
            }
#pragma unroll
            for (int d = 1; d < 16; d <<= 1) rsum += __shfl_xor(rsum, d, 32);
            li[j] = li[j] * alpha + rsum;
#pragma unroll
            for (int ct = 0; ct < 4; ++ct) accO[ct][j] *= alpha;
            const int row = j + 8 * hl;
#pragma unroll
            for (int st = 0; st < 8; ++st)
                Pb[wv][row][st * 16 + ln] = (_Float16)S[st][j];
        }
        asm volatile("" ::: "memory");   // order P stash vs b128 re-read (same wave)

        // ---- O += P * V ----
#pragma unroll
        for (int kk = 0; kk < 4; ++kk) {
            const v16h ap = load_fA_k(&Pb[wv][0][kk * 32], 128, lane);
#pragma unroll
            for (int ct = 0; ct < 4; ++ct) {
                const v16h bv = load_fB_k(&Vt[ct * 16][kk * 32], 128, lane);
                accO[ct] = wmma16x16x32(ap, bv, accO[ct]);
            }
        }
        __syncthreads();   // before tiles are overwritten
    }

    // ---- finalize: ah[b][t][h*64 + c] = O / l  (c contiguous -> coalesced) --
    const int b = bh >> 3, h = bh & 7;
#pragma unroll
    for (int ct = 0; ct < 4; ++ct)
#pragma unroll
        for (int j = 0; j < 8; ++j) {
            const int t = m0 + j + 8 * hl;
            const int c = h * CH_ + ct * 16 + ln;
            ah[((size_t)b * T_ + t) * C_ + c] = (_Float16)(accO[ct][j] / li[j]);
        }
}

// ------------- kernel 4: proj GEMM + bias + residual ----------------
__global__ __launch_bounds__(128) void proj_gemm(const _Float16* __restrict__ ah,
                                                 const _Float16* __restrict__ wph,
                                                 const float* __restrict__ bp,
                                                 const float* __restrict__ x,
                                                 float* __restrict__ out) {
    const int lane = threadIdx.x & 31;
    const int wv   = threadIdx.x >> 5;
    const int b    = blockIdx.z;
    const int m0   = blockIdx.y * 64 + wv * 16;   // o base
    const int n0   = blockIdx.x * 64;             // t base
    const _Float16* A  = wph + (size_t)m0 * C_;               // [o][c]
    const _Float16* Bp = ah + ((size_t)b * T_ + n0) * C_;     // [t][c]
    v8f acc[4] = {};
#pragma unroll 4
    for (int k = 0; k < C_; k += 32) {
        const v16h a = load_fA_k(A + k, C_, lane);
#pragma unroll
        for (int nt = 0; nt < 4; ++nt) {
            const v16h bb = load_fB_k(Bp + (size_t)(nt * 16) * C_ + k, C_, lane);
            acc[nt] = wmma16x16x32(a, bb, acc[nt]);
        }
    }
    const int hl = lane >> 4, ln = lane & 15;
#pragma unroll
    for (int nt = 0; nt < 4; ++nt)
#pragma unroll
        for (int j = 0; j < 8; ++j) {
            const int o = m0 + j + 8 * hl;
            const int t = n0 + nt * 16 + ln;
            const size_t idx = ((size_t)b * C_ + o) * T_ + t;
            out[idx] = x[idx] + acc[nt][j] + bp[o];
        }
}

// ---------------- launch ----------------
extern "C" void kernel_launch(void* const* d_in, const int* in_sizes, int n_in,
                              void* d_out, int out_size, void* d_ws, size_t ws_size,
                              hipStream_t stream) {
    const float* x  = (const float*)d_in[0];
    const float* gw = (const float*)d_in[2];
    const float* gb = (const float*)d_in[3];
    const float* wq = (const float*)d_in[4];
    const float* bq = (const float*)d_in[5];
    const float* wp = (const float*)d_in[6];
    const float* bp = (const float*)d_in[7];
    float* out = (float*)d_out;
    char* ws = (char*)d_ws;

    _Float16* xn  = (_Float16*)(ws + XN_OFF);
    _Float16* qt  = (_Float16*)(ws + QT_OFF);
    _Float16* kt  = (_Float16*)(ws + KT_OFF);
    _Float16* vt  = (_Float16*)(ws + VT_OFF);
    _Float16* ah  = (_Float16*)(ws + AH_OFF);
    _Float16* wqh = (_Float16*)(ws + WQH_OFF);
    _Float16* wph = (_Float16*)(ws + WPH_OFF);

    cvt_weights<<<(O3C_ * C_ + 255) / 256, 256, 0, stream>>>(wq, wp, wqh, wph);
    groupnorm<<<B_ * GROUPS_, 256, 0, stream>>>(x, gw, gb, xn);
    qkv_gemm<<<dim3(T_ / 64, O3C_ / 64, B_), 128, 0, stream>>>(xn, wqh, bq, qt, kt, vt);
    flash_attn<<<dim3(T_ / 64, B_ * HEADS_), 128, 0, stream>>>(qt, kt, vt, ah);
    proj_gemm<<<dim3(T_ / 64, C_ / 64, B_), 128, 0, stream>>>(ah, wph, bp, x, out);
}
